// server_hgcn_21964462752541
// MI455X (gfx1250) — matrix operand order repaired and checked by reading, not compiled, not run on hardware
//
#include <hip/hip_runtime.h>
#include <hip/hip_bf16.h>

typedef __attribute__((ext_vector_type(16))) _Float16 v16h;
typedef __attribute__((ext_vector_type(8)))  float    v8f;

#define N_NODES 100000
#define E_EDGES 1600000
#define NF 128
#define NH 64
#define NC 16
#define EPSF 1e-15f
#define MAXN 0.996f   /* (1 - 4e-3)/sqrt(c), c=1 */

__device__ __forceinline__ float artanh_c(float x) {
    x = fminf(fmaxf(x, -1.f + 1e-7f), 1.f - 1e-7f);
    return 0.5f * logf((1.f + x) / (1.f - x));
}

// ---------------------------------------------------------------------------
// hyp_b = proj(expmap0(b)) for b1,b2,b3 (each len 64) + squared norms. 1 block of 64.
__global__ void k_hyp_b(const float* __restrict__ b1, const float* __restrict__ b2,
                        const float* __restrict__ b3, float* __restrict__ hb,
                        float* __restrict__ hb2) {
    __shared__ float red[64];
    const float* bs[3] = {b1, b2, b3};
    int t = threadIdx.x;
    for (int i = 0; i < 3; ++i) {
        float v = bs[i][t];
        red[t] = v * v;
        __syncthreads();
        if (t == 0) { float s = 0.f; for (int k = 0; k < 64; ++k) s += red[k]; red[0] = s; }
        __syncthreads();
        float ss = red[0];
        __syncthreads();
        float n = fmaxf(sqrtf(ss), EPSF);
        float s = tanhf(n) / n;
        float rn = s * n;
        if (rn > MAXN) s *= MAXN / rn;
        float hv = s * v;
        hb[i * 64 + t] = hv;
        red[t] = hv * hv;
        __syncthreads();
        if (t == 0) { float s2 = 0.f; for (int k = 0; k < 64; ++k) s2 += red[k]; hb2[i] = s2; }
        __syncthreads();
    }
}

// ---------------------------------------------------------------------------
// h0 = proj(expmap0(x)), [N,128]. One wave per row, 4 floats/lane.
__global__ __launch_bounds__(256) void k_expmap0(const float* __restrict__ x,
                                                 float* __restrict__ h, int nrows) {
    int tid = threadIdx.x, w = tid >> 5, L = tid & 31;
    int r = blockIdx.x * 8 + w;
    int rc = r < nrows ? r : nrows - 1;
    const float* xp = x + (size_t)rc * NF;
    float4 q = *(const float4*)(xp + L * 4);
    float ss = q.x * q.x + q.y * q.y + q.z * q.z + q.w * q.w;
    for (int m = 1; m < 32; m <<= 1) ss += __shfl_xor(ss, m);
    float n = fmaxf(sqrtf(ss), EPSF);
    float s = tanhf(n) / n;
    float rn = s * n;
    s = (rn > MAXN) ? s * (MAXN / rn) : s;
    if (r < nrows) {
        float4 o; o.x = s * q.x; o.y = s * q.y; o.z = s * q.z; o.w = s * q.w;
        *(float4*)(h + (size_t)r * NF + L * 4) = o;
    }
}

// ---------------------------------------------------------------------------
// Fused HypLinear via WMMA:  ht[r,:] = u(r)*mp[r,:] + v(r)*hyp_b
// mp = p @ W^T; scalars u,v implement mobius_matvec+proj+mobius_add+proj+logmap0.
// Block = 8 waves; wave handles 16 rows x 64 cols.
// B fragments are pre-swizzled in LDS into WMMA layout: one contiguous v16h
// per (ktile, otile, lane) -> a single 32B LDS vector load per WMMA operand.
template <int IN>
__global__ __launch_bounds__(256) void k_linear(const float* __restrict__ in,
                                                const float* __restrict__ W,
                                                const float* __restrict__ hb,
                                                const float* __restrict__ hb2p,
                                                float* __restrict__ ht, int nrows) {
    constexpr int OUT = 64;
    constexpr int KT = IN / 32;
    __shared__ v16h  sWf[KT * 4 * 32];      // B fragments, fragment-major
    __shared__ float sMp[8 * 16 * 64];      // per-wave mp tiles
    __shared__ float sPs[8 * 32];           // per-lane input sumsq halves
    __shared__ float sHb[64];

    const int tid = threadIdx.x;
    const int w = tid >> 5;
    const int L = tid & 31;
    const int half = L >> 4;
    const int l16 = L & 15;

    // Swizzle W^T into fragment layout:
    // frag f = ((kt*4+o)*32+lane)*16 + j  holds  W[(o*16 + lane%16) * IN + kt*32 + (lane/16)*8 + (j%8) + (j/8)*16]
    {
        _Float16* sWfH = (_Float16*)sWf;
        for (int f = tid; f < IN * OUT; f += 256) {
            int j  = f & 15;
            int ln = (f >> 4) & 31;
            int o  = (f >> 9) & 3;
            int kt = f >> 11;
            int k   = kt * 32 + (ln >> 4) * 8 + (j & 7) + (j >> 3) * 16;
            int col = o * 16 + (ln & 15);
            sWfH[f] = (_Float16)W[col * IN + k];
        }
    }
    if (tid < 64) sHb[tid] = hb[tid];
    __syncthreads();
    const float hb2 = *hb2p;

    const int m0 = blockIdx.x * 128 + w * 16;
    int gr = m0 + l16;
    if (gr >= nrows) gr = nrows - 1;
    const float* rp = in + (size_t)gr * IN;

    // A fragments (16-bit A layout: lane<16 -> K {0..7,16..23}, lane>=16 -> {8..15,24..31})
    v16h afr[KT];
    float psq = 0.f;
#pragma unroll
    for (int kt = 0; kt < KT; ++kt) {
        int K0 = kt * 32 + half * 8;
        float4 q0 = *(const float4*)(rp + K0);
        float4 q1 = *(const float4*)(rp + K0 + 4);
        float4 q2 = *(const float4*)(rp + K0 + 16);
        float4 q3 = *(const float4*)(rp + K0 + 20);
        psq += q0.x*q0.x + q0.y*q0.y + q0.z*q0.z + q0.w*q0.w
             + q1.x*q1.x + q1.y*q1.y + q1.z*q1.z + q1.w*q1.w
             + q2.x*q2.x + q2.y*q2.y + q2.z*q2.z + q2.w*q2.w
             + q3.x*q3.x + q3.y*q3.y + q3.z*q3.z + q3.w*q3.w;
        v16h a;
        a[0]=(_Float16)q0.x; a[1]=(_Float16)q0.y; a[2]=(_Float16)q0.z; a[3]=(_Float16)q0.w;
        a[4]=(_Float16)q1.x; a[5]=(_Float16)q1.y; a[6]=(_Float16)q1.z; a[7]=(_Float16)q1.w;
        a[8]=(_Float16)q2.x; a[9]=(_Float16)q2.y; a[10]=(_Float16)q2.z; a[11]=(_Float16)q2.w;
        a[12]=(_Float16)q3.x; a[13]=(_Float16)q3.y; a[14]=(_Float16)q3.z; a[15]=(_Float16)q3.w;
        afr[kt] = a;
    }
    sPs[w * 32 + L] = psq;

#pragma unroll
    for (int o = 0; o < 4; ++o) {
        v8f acc = {};
#pragma unroll
        for (int kt = 0; kt < KT; ++kt) {
            v16h b = sWf[(kt * 4 + o) * 32 + L];   // 2x ds_load_b128
            acc = __builtin_amdgcn_wmma_f32_16x16x32_f16(false, afr[kt], false, b,
                                                         (short)0, acc, false, false);
        }
#pragma unroll
        for (int v = 0; v < 8; ++v)
            sMp[w * 1024 + (v + half * 8) * 64 + o * 16 + l16] = acc[v];
    }
    __syncthreads();

    // Epilogue: 2 lanes per row, 32 cols each (vectorized LDS reads).
    const int r16 = L >> 1;
    const int cb = (L & 1) * 32;
    const float* mprow = &sMp[w * 1024 + r16 * 64 + cb];
    float ss = 0.f, dh = 0.f;
#pragma unroll
    for (int c = 0; c < 32; c += 4) {
        float4 m  = *(const float4*)(mprow + c);
        float4 hv = *(const float4*)(&sHb[cb + c]);
        ss += m.x*m.x + m.y*m.y + m.z*m.z + m.w*m.w;
        dh += m.x*hv.x + m.y*hv.y + m.z*hv.z + m.w*hv.w;
    }
    ss += __shfl_xor(ss, 1);
    dh += __shfl_xor(dh, 1);
    float psqr = sPs[w * 32 + r16] + sPs[w * 32 + r16 + 16];

    float pn  = fmaxf(sqrtf(psqr), EPSF);
    float mpn = sqrtf(ss);
    float s = tanhf(mpn / pn * artanh_c(pn)) / fmaxf(mpn, EPSF);
    s = (mpn <= EPSF) ? 0.f : s;
    float rn = s * mpn;                                  // proj(res)
    s = (rn > MAXN) ? s * (MAXN / rn) : s;
    float x2 = s * s * ss, xy = s * dh;                  // mobius_add(res, hyp_b)
    float ax = 1.f + 2.f * xy + hb2;
    float ay = 1.f - x2;
    float den = fmaxf(1.f + 2.f * xy + x2 * hb2, EPSF);
    float u = ax * s / den, v = ay / den;
    float on2 = u * u * ss + 2.f * u * v * dh + v * v * hb2;   // proj
    float on = sqrtf(fmaxf(on2, 0.f));
    float k2 = (on > MAXN) ? (MAXN / on) : 1.f;
    u *= k2; v *= k2; on = fminf(on, MAXN);
    float n = fmaxf(on, EPSF);                           // logmap0
    float f = artanh_c(n) / n;
    u *= f; v *= f;

    int orow = m0 + r16;
    if (orow < nrows) {
        float* op = ht + (size_t)orow * 64 + cb;
#pragma unroll
        for (int c = 0; c < 32; c += 4) {
            float4 m  = *(const float4*)(mprow + c);
            float4 hv = *(const float4*)(&sHb[cb + c]);
            float4 r;
            r.x = u * m.x + v * hv.x; r.y = u * m.y + v * hv.y;
            r.z = u * m.z + v * hv.z; r.w = u * m.w + v * hv.w;
            *(float4*)(op + c) = r;
        }
    }
}

// ---------------------------------------------------------------------------
__global__ __launch_bounds__(256) void k_zero(float* __restrict__ p, int n) {
    int i = blockIdx.x * 256 + threadIdx.x;
    if (i < n) p[i] = 0.f;
}

// agg[dst] += adj * ht[src]; one wave per edge, 2 feats/lane.
__global__ __launch_bounds__(256) void k_agg(const int* __restrict__ src,
                                             const int* __restrict__ dst,
                                             const float* __restrict__ adj,
                                             const float* __restrict__ ht,
                                             float* __restrict__ agg, int nedges) {
    long idx = (long)blockIdx.x * 256 + threadIdx.x;
    int e = (int)(idx >> 5);
    int c0 = ((int)idx & 31) * 2;
    if (e < nedges) {
        int s = src[e], d = dst[e];
        float a = adj[e];
        float2 v = *(const float2*)(ht + (size_t)s * 64 + c0);
        atomicAdd(agg + (size_t)d * 64 + c0, a * v.x);
        atomicAdd(agg + (size_t)d * 64 + c0 + 1, a * v.y);
    }
}

// h = proj(expmap0(relu(logmap0(proj(expmap0(agg)))))). One wave per row.
__global__ __launch_bounds__(256) void k_postagg(const float* __restrict__ agg,
                                                 float* __restrict__ h, int nrows) {
    int tid = threadIdx.x, w = tid >> 5, L = tid & 31;
    int r = blockIdx.x * 8 + w;
    int rc = r < nrows ? r : nrows - 1;
    float2 g = *(const float2*)(agg + (size_t)rc * 64 + L * 2);
    float ss = g.x * g.x + g.y * g.y;
    for (int m = 1; m < 32; m <<= 1) ss += __shfl_xor(ss, m);
    float n = fmaxf(sqrtf(ss), EPSF);
    float s = tanhf(n) / n;
    float rn = s * n;
    s = (rn > MAXN) ? s * (MAXN / rn) : s;
    float ln = fmaxf(s * n, EPSF);
    float fl = artanh_c(ln) / ln;
    float tx = fmaxf(fl * s * g.x, 0.f);
    float ty = fmaxf(fl * s * g.y, 0.f);
    float tss = tx * tx + ty * ty;
    for (int m = 1; m < 32; m <<= 1) tss += __shfl_xor(tss, m);
    float tn = fmaxf(sqrtf(tss), EPSF);
    float s2 = tanhf(tn) / tn;
    float rn2 = s2 * tn;
    s2 = (rn2 > MAXN) ? s2 * (MAXN / rn2) : s2;
    if (r < nrows) {
        float2 o; o.x = s2 * tx; o.y = s2 * ty;
        *(float2*)(h + (size_t)r * 64 + L * 2) = o;
    }
}

// ---------------------------------------------------------------------------
// Head: t=logmap0(h); z=relu(t@W4^T+b4); out=log_softmax(z@W5^T+b5). Wave/row.
__global__ __launch_bounds__(256) void k_head(const float* __restrict__ h,
                                              const float* __restrict__ W4,
                                              const float* __restrict__ b4,
                                              const float* __restrict__ W5,
                                              const float* __restrict__ b5,
                                              float* __restrict__ out, int nrows) {
    __shared__ float sW4[32 * 64], sW5[16 * 32], sB4[32], sB5[16];
    __shared__ float sT[8 * 64], sZ[8 * 32];
    int tid = threadIdx.x, w = tid >> 5, L = tid & 31;
    for (int i = tid; i < 32 * 64; i += 256) sW4[i] = W4[i];
    for (int i = tid; i < 16 * 32; i += 256) sW5[i] = W5[i];
    if (tid < 32) sB4[tid] = b4[tid];
    if (tid < 16) sB5[tid] = b5[tid];
    __syncthreads();

    int r = blockIdx.x * 8 + w;
    int rc = r < nrows ? r : nrows - 1;
    float2 g = *(const float2*)(h + (size_t)rc * 64 + L * 2);
    float ss = g.x * g.x + g.y * g.y;
    for (int m = 1; m < 32; m <<= 1) ss += __shfl_xor(ss, m);
    float n = fmaxf(sqrtf(ss), EPSF);
    float f = artanh_c(n) / n;
    sT[w * 64 + L * 2] = f * g.x;
    sT[w * 64 + L * 2 + 1] = f * g.y;
    __syncthreads();

    float z = sB4[L];
#pragma unroll
    for (int k = 0; k < 64; k += 4) {
        float4 wv = *(const float4*)(&sW4[L * 64 + k]);
        float4 tv = *(const float4*)(&sT[w * 64 + k]);
        z += tv.x * wv.x + tv.y * wv.y + tv.z * wv.z + tv.w * wv.w;
    }
    z = fmaxf(z, 0.f);
    sZ[w * 32 + L] = z;
    __syncthreads();

    int j = L & 15;
    float lg = sB5[j];
#pragma unroll
    for (int k = 0; k < 32; k += 4) {
        float4 wv = *(const float4*)(&sW5[j * 32 + k]);
        float4 zv = *(const float4*)(&sZ[w * 32 + k]);
        lg += zv.x * wv.x + zv.y * wv.y + zv.z * wv.z + zv.w * wv.w;
    }
    float mx = lg;
    for (int m = 1; m < 16; m <<= 1) mx = fmaxf(mx, __shfl_xor(mx, m, 16));
    float se = expf(lg - mx);
    for (int m = 1; m < 16; m <<= 1) se += __shfl_xor(se, m, 16);
    if (r < nrows && L < 16) out[(size_t)r * 16 + j] = lg - mx - logf(se);
}

// ---------------------------------------------------------------------------
extern "C" void kernel_launch(void* const* d_in, const int* in_sizes, int n_in,
                              void* d_out, int out_size, void* d_ws, size_t ws_size,
                              hipStream_t stream) {
    const float* x   = (const float*)d_in[0];
    const int*   src = (const int*)d_in[1];
    const int*   dst = (const int*)d_in[2];
    const float* adj = (const float*)d_in[3];
    const float* W1  = (const float*)d_in[4];
    const float* b1  = (const float*)d_in[5];
    const float* W2  = (const float*)d_in[6];
    const float* b2  = (const float*)d_in[7];
    const float* W3  = (const float*)d_in[8];
    const float* b3  = (const float*)d_in[9];
    const float* W4  = (const float*)d_in[10];
    const float* b4  = (const float*)d_in[11];
    const float* W5  = (const float*)d_in[12];
    const float* b5  = (const float*)d_in[13];
    float* out = (float*)d_out;

    const int N = N_NODES, E = E_EDGES;
    float* ws   = (float*)d_ws;
    float* H128 = ws;                       // [N,128]
    float* HT   = H128 + (size_t)N * 128;   // [N,64]
    float* AGG  = HT + (size_t)N * 64;      // [N,64]
    float* H64  = AGG + (size_t)N * 64;     // [N,64]
    float* HB   = H64 + (size_t)N * 64;     // [3,64]
    float* HB2  = HB + 192;                 // [3]

    const int rowBlk8 = (N + 7) / 8;        // wave-per-row kernels
    const int linBlk  = (N + 127) / 128;    // 16 rows per wave, 8 waves
    const int zeroBlk = (N * 64 + 255) / 256;
    const long aggThreads = (long)E * 32;
    const int aggBlk = (int)((aggThreads + 255) / 256);

    k_hyp_b<<<1, 64, 0, stream>>>(b1, b2, b3, HB, HB2);
    k_expmap0<<<rowBlk8, 256, 0, stream>>>(x, H128, N);

    // layer 1
    k_linear<128><<<linBlk, 256, 0, stream>>>(H128, W1, HB + 0, HB2 + 0, HT, N);
    k_zero<<<zeroBlk, 256, 0, stream>>>(AGG, N * 64);
    k_agg<<<aggBlk, 256, 0, stream>>>(src, dst, adj, HT, AGG, E);
    k_postagg<<<rowBlk8, 256, 0, stream>>>(AGG, H64, N);

    // layer 2
    k_linear<64><<<linBlk, 256, 0, stream>>>(H64, W2, HB + 64, HB2 + 1, HT, N);
    k_zero<<<zeroBlk, 256, 0, stream>>>(AGG, N * 64);
    k_agg<<<aggBlk, 256, 0, stream>>>(src, dst, adj, HT, AGG, E);
    k_postagg<<<rowBlk8, 256, 0, stream>>>(AGG, H64, N);

    // layer 3
    k_linear<64><<<linBlk, 256, 0, stream>>>(H64, W3, HB + 128, HB2 + 2, HT, N);
    k_zero<<<zeroBlk, 256, 0, stream>>>(AGG, N * 64);
    k_agg<<<aggBlk, 256, 0, stream>>>(src, dst, adj, HT, AGG, E);
    k_postagg<<<rowBlk8, 256, 0, stream>>>(AGG, H64, N);

    k_head<<<rowBlk8, 256, 0, stream>>>(H64, W4, b4, W5, b5, out, N);
}
